// DomainAdaptationLayer_64862596104324
// MI455X (gfx1250) — matrix operand order repaired and checked by reading, not compiled, not run on hardware
//
#include <hip/hip_runtime.h>
#include <hip/hip_bf16.h>
#include <math.h>

// Problem constants (from reference setup_inputs)
#define BB 4096
#define FF 1024
#define UU 1024
#define GG 16
#define MM 64
#define INV2S2 2.0f   // 1/(2*0.5^2)

typedef __attribute__((ext_vector_type(16))) __bf16 v16bf;
typedef __attribute__((ext_vector_type(8)))  __bf16 v8bf;
typedef __attribute__((ext_vector_type(4)))  __bf16 v4bf;
typedef __attribute__((ext_vector_type(8)))  float  v8f;

// ---------------------------------------------------------------------------
// CDNA5 ISA 7.12.2 bf16 WMMA fragment layouts (wave32):
//  A 16x32: lane L holds row M=L%16; elements 0..7  = K[8h .. 8h+7],
//           elements 8..15 = K[16+8h .. 16+8h+7]        (h = L/16)
//  B 32x16: lane L holds col N=L%16; same k order as A; code uses elems 0..15 = K[16h .. 16h+15] for A and B
//  C/D 16x16 f32: VGPR i, lane L -> M = i + 8*(L/16), N = L%16
// With bf16 K-contiguous memory, each fragment is two 16-byte loads.
// ---------------------------------------------------------------------------

__device__ __forceinline__ v16bf load_a_bf16(const __bf16* __restrict__ row,
                                             int k0, int half) {
  v8bf lo = *(const v8bf*)(row + k0 + 16 * half);       // K 16h..16h+7
  v8bf hi = *(const v8bf*)(row + k0 + 16 * half + 8);   // K 16h+8..16h+15
  return __builtin_shufflevector(lo, hi, 0, 1, 2, 3, 4, 5, 6, 7,
                                 8, 9, 10, 11, 12, 13, 14, 15);
}

__device__ __forceinline__ v16bf load_b_bf16(const __bf16* __restrict__ row,
                                             int k0, int half) {
  v8bf lo = *(const v8bf*)(row + k0 + 16 * half);       // K 16h..16h+7
  v8bf hi = *(const v8bf*)(row + k0 + 16 * half + 8);   // K 16h+8..16h+15
  return __builtin_shufflevector(lo, hi, 0, 1, 2, 3, 4, 5, 6, 7,
                                 8, 9, 10, 11, 12, 13, 14, 15);
}

// ---- f32 fallback fragment loaders (used only if ws is too small) ---------
__device__ __forceinline__ v16bf load_a_f32(const float* __restrict__ row,
                                            int k0, int half) {
  v16bf a;
#pragma unroll
  for (int p = 0; p < 8; ++p) {
    int k = 2 * p + 16 * half;
    const float2 v = *(const float2*)(row + k0 + k);
    a[2 * p] = (__bf16)v.x;
    a[2 * p + 1] = (__bf16)v.y;
  }
  return a;
}

__device__ __forceinline__ v16bf load_b_f32_kcontig(const float* __restrict__ row,
                                                    int k0, int half) {
  v16bf b;
  const float4* p4 = (const float4*)(row + k0 + 16 * half);
#pragma unroll
  for (int q = 0; q < 4; ++q) {
    float4 v = p4[q];
    b[4 * q + 0] = (__bf16)v.x;
    b[4 * q + 1] = (__bf16)v.y;
    b[4 * q + 2] = (__bf16)v.z;
    b[4 * q + 3] = (__bf16)v.w;
  }
  return b;
}

__device__ __forceinline__ v16bf load_b_f32_kstrided(const float* __restrict__ Wg,
                                                     int k0, int col, int half) {
  v16bf b;
#pragma unroll
  for (int v = 0; v < 8; ++v) {
    int k = k0 + 2 * v + 16 * half;
    b[2 * v] = (__bf16)Wg[(size_t)k * UU + col];
    b[2 * v + 1] = (__bf16)Wg[(size_t)(k + 1) * UU + col];
  }
  return b;
}

// ---------------------------------------------------------------------------
// Kernel: h2[b] = sum_f h[b,f]^2   (one wave per row)
// ---------------------------------------------------------------------------
__global__ void k_h2(const float* __restrict__ h, float* __restrict__ h2) {
  int wave = threadIdx.x >> 5, lane = threadIdx.x & 31;
  int row = blockIdx.x * 8 + wave;
  const float* r = h + (size_t)row * FF;
  float s = 0.f;
  for (int i = lane; i < FF; i += 32) {
    float v = r[i];
    s += v * v;
  }
#pragma unroll
  for (int off = 16; off; off >>= 1) s += __shfl_xor(s, off, 32);
  if (lane == 0) h2[row] = s;
}

// ---------------------------------------------------------------------------
// Kernel: per group g: d2[g,m] and mean_Kdd[g]  (one block per g)
// ---------------------------------------------------------------------------
__global__ void k_dstats(const float* __restrict__ D, float* __restrict__ d2,
                         float* __restrict__ mkdd) {
  __shared__ float sd2[MM];
  __shared__ float ssum;
  int g = blockIdx.x;
  int t = threadIdx.x;
  const float* Dg = D + (size_t)g * MM * FF;
  if (t < MM) sd2[t] = 0.f;
  if (t == 0) ssum = 0.f;
  __syncthreads();
  {
    int m = t & (MM - 1), c = t >> 6;
    const float4* row = (const float4*)(Dg + (size_t)m * FF) + c * 64;
    float p = 0.f;
    for (int i = 0; i < 64; ++i) {
      float4 v = row[i];
      p += v.x * v.x + v.y * v.y + v.z * v.z + v.w * v.w;
    }
    atomicAdd(&sd2[m], p);
  }
  __syncthreads();
  if (t < MM) d2[g * MM + t] = sd2[t];
  float local = 0.f;
  for (int p = t; p < MM * MM; p += 256) {
    int m = p >> 6, n = p & (MM - 1);
    const float4* rm = (const float4*)(Dg + (size_t)m * FF);
    const float4* rn = (const float4*)(Dg + (size_t)n * FF);
    float dot = 0.f;
    for (int i = 0; i < FF / 4; ++i) {
      float4 a = rm[i], b = rn[i];
      dot += a.x * b.x + a.y * b.y + a.z * b.z + a.w * b.w;
    }
    local += expf(-(sd2[m] - 2.f * dot + sd2[n]) * INV2S2);
  }
  atomicAdd(&ssum, local);
  __syncthreads();
  if (t == 0) mkdd[g] = ssum / (float)(MM * MM);
}

// ---------------------------------------------------------------------------
// Conversion pre-passes (fast path): f32 -> bf16 (RNE)
// ---------------------------------------------------------------------------
__global__ void k_cvt_bf16(const float* __restrict__ src, __bf16* __restrict__ dst) {
  size_t i = ((size_t)blockIdx.x * blockDim.x + threadIdx.x) * 4;
  float4 v = *(const float4*)(src + i);
  v4bf o;
  o[0] = (__bf16)v.x; o[1] = (__bf16)v.y; o[2] = (__bf16)v.z; o[3] = (__bf16)v.w;
  *(v4bf*)(dst + i) = o;
}

// W[g][f][u] f32  ->  Wt[g][u][f] bf16  (LDS-tiled transpose, 32x32)
__global__ void k_wt(const float* __restrict__ W, __bf16* __restrict__ Wt) {
  __shared__ float tile[32][33];
  int g = blockIdx.z;
  int u0 = blockIdx.x * 32, f0 = blockIdx.y * 32;
  const float* Wg = W + (size_t)g * FF * UU;
  __bf16* Wtg = Wt + (size_t)g * UU * FF;
  int tx = threadIdx.x & 31, ty = threadIdx.x >> 5;  // 32 x 8
#pragma unroll
  for (int r = 0; r < 4; ++r)
    tile[ty + 8 * r][tx] = Wg[(size_t)(f0 + ty + 8 * r) * UU + (u0 + tx)];
  __syncthreads();
#pragma unroll
  for (int r = 0; r < 4; ++r)
    Wtg[(size_t)(u0 + ty + 8 * r) * FF + (f0 + tx)] = (__bf16)tile[tx][ty + 8 * r];
}

// ---------------------------------------------------------------------------
// Kernel: mean_Khd[g,b] via bf16 WMMA (h @ D_g^T) fused with exp/mean.
// grid (B/128, G), 256 threads; wave owns 16 rows x all 64 m.
// ---------------------------------------------------------------------------
__global__ void k_meankhd_bf16(const __bf16* __restrict__ hb, const __bf16* __restrict__ Db,
                               const float* __restrict__ h2, const float* __restrict__ d2,
                               float* __restrict__ mkhd) {
  int g = blockIdx.y;
  int lane = threadIdx.x & 31, wave = threadIdx.x >> 5;
  int rowbase = blockIdx.x * 128 + wave * 16;
  int half = lane >> 4, n = lane & 15;
  const __bf16* hrow = hb + (size_t)(rowbase + n) * FF;
  const __bf16* Dg = Db + (size_t)g * MM * FF;

  v8f acc[4] = {};
  for (int k0 = 0; k0 < FF; k0 += 32) {
    v16bf a = load_a_bf16(hrow, k0, half);
#pragma unroll
    for (int j = 0; j < 4; ++j) {
      const __bf16* drow = Dg + (size_t)(j * 16 + n) * FF;  // B col = m
      v16bf bf = load_b_bf16(drow, k0, half);
      acc[j] = __builtin_amdgcn_wmma_f32_16x16x32_bf16(false, a, false, bf,
                                                       (short)0, acc[j], false, false);
    }
  }
  float d2v[4];
#pragma unroll
  for (int j = 0; j < 4; ++j) d2v[j] = d2[g * MM + j * 16 + n];
#pragma unroll
  for (int i = 0; i < 8; ++i) {
    int row = rowbase + i + 8 * half;
    float hh = h2[row];
    float s = 0.f;
#pragma unroll
    for (int j = 0; j < 4; ++j)
      s += expf(-(hh - 2.f * acc[j][i] + d2v[j]) * INV2S2);
#pragma unroll
    for (int off = 1; off < 16; off <<= 1) s += __shfl_xor(s, off, 32);
    if (n == 0) mkhd[(size_t)g * BB + row] = s * (1.f / (float)MM);
  }
}

// f32 fallback variant
__global__ void k_meankhd_f32(const float* __restrict__ h, const float* __restrict__ D,
                              const float* __restrict__ h2, const float* __restrict__ d2,
                              float* __restrict__ mkhd) {
  int g = blockIdx.y;
  int lane = threadIdx.x & 31, wave = threadIdx.x >> 5;
  int rowbase = blockIdx.x * 128 + wave * 16;
  int half = lane >> 4, n = lane & 15;
  const float* hrow = h + (size_t)(rowbase + n) * FF;
  const float* Dg = D + (size_t)g * MM * FF;
  v8f acc[4] = {};
  for (int k0 = 0; k0 < FF; k0 += 32) {
    v16bf a = load_a_f32(hrow, k0, half);
#pragma unroll
    for (int j = 0; j < 4; ++j) {
      const float* drow = Dg + (size_t)(j * 16 + n) * FF;
      v16bf bf = load_b_f32_kcontig(drow, k0, half);
      acc[j] = __builtin_amdgcn_wmma_f32_16x16x32_bf16(false, a, false, bf,
                                                       (short)0, acc[j], false, false);
    }
  }
  float d2v[4];
#pragma unroll
  for (int j = 0; j < 4; ++j) d2v[j] = d2[g * MM + j * 16 + n];
#pragma unroll
  for (int i = 0; i < 8; ++i) {
    int row = rowbase + i + 8 * half;
    float hh = h2[row];
    float s = 0.f;
#pragma unroll
    for (int j = 0; j < 4; ++j)
      s += expf(-(hh - 2.f * acc[j][i] + d2v[j]) * INV2S2);
#pragma unroll
    for (int off = 1; off < 16; off <<= 1) s += __shfl_xor(s, off, 32);
    if (n == 0) mkhd[(size_t)g * BB + row] = s * (1.f / (float)MM);
  }
}

// ---------------------------------------------------------------------------
// Kernel: prob[g,b] = softmax_g(-(1 - 2*mean_Khd + mean_Kdd))
// ---------------------------------------------------------------------------
__global__ void k_prob(const float* __restrict__ mkhd, const float* __restrict__ mkdd,
                       float* __restrict__ prob) {
  int b = blockIdx.x * blockDim.x + threadIdx.x;
  float v[GG];
  float mx = -3.4e38f;
#pragma unroll
  for (int g = 0; g < GG; ++g) {
    float mmd = 1.f - 2.f * mkhd[(size_t)g * BB + b] + mkdd[g];
    v[g] = -mmd;  // SOFTNESS = 1.0
    mx = fmaxf(mx, v[g]);
  }
  float sum = 0.f;
#pragma unroll
  for (int g = 0; g < GG; ++g) {
    v[g] = expf(v[g] - mx);
    sum += v[g];
  }
  float inv = 1.f / sum;
#pragma unroll
  for (int g = 0; g < GG; ++g) prob[(size_t)g * BB + b] = v[g] * inv;
}

// ---------------------------------------------------------------------------
// Main GEMM (fast path): out[b,u] = sum_g prob[g,b] * (h @ W_g + bias_g)[b,u]
// hb: bf16 [B][F];  Wt: bf16 [G][U][F] (K contiguous per output column)
// grid (B/128, U/64), 256 threads; wave owns 16x64 tile.
// ---------------------------------------------------------------------------
__global__ void k_main_bf16(const __bf16* __restrict__ hb, const __bf16* __restrict__ Wt,
                            const float* __restrict__ bias, const float* __restrict__ prob,
                            float* __restrict__ out) {
  int lane = threadIdx.x & 31, wave = threadIdx.x >> 5;
  int rowbase = blockIdx.x * 128 + wave * 16;
  int colbase = blockIdx.y * 64;
  int half = lane >> 4, n = lane & 15;
  const __bf16* hrow = hb + (size_t)(rowbase + n) * FF;

  // per-lane column rows of Wt for the 4 j-tiles
  const __bf16* brow0 = Wt + (size_t)(colbase + 0 * 16 + n) * FF;
  const __bf16* brow1 = Wt + (size_t)(colbase + 1 * 16 + n) * FF;
  const __bf16* brow2 = Wt + (size_t)(colbase + 2 * 16 + n) * FF;
  const __bf16* brow3 = Wt + (size_t)(colbase + 3 * 16 + n) * FF;

  v8f facc[4] = {};
  for (int g = 0; g < GG; ++g) {
    size_t goff = (size_t)g * UU * FF;
    v8f gacc[4] = {};
    for (int k0 = 0; k0 < FF; k0 += 32) {
      v16bf a = load_a_bf16(hrow, k0, half);
      v16bf b0 = load_b_bf16(brow0 + goff, k0, half);
      v16bf b1 = load_b_bf16(brow1 + goff, k0, half);
      v16bf b2 = load_b_bf16(brow2 + goff, k0, half);
      v16bf b3 = load_b_bf16(brow3 + goff, k0, half);
      gacc[0] = __builtin_amdgcn_wmma_f32_16x16x32_bf16(false, a, false, b0,
                                                        (short)0, gacc[0], false, false);
      gacc[1] = __builtin_amdgcn_wmma_f32_16x16x32_bf16(false, a, false, b1,
                                                        (short)0, gacc[1], false, false);
      gacc[2] = __builtin_amdgcn_wmma_f32_16x16x32_bf16(false, a, false, b2,
                                                        (short)0, gacc[2], false, false);
      gacc[3] = __builtin_amdgcn_wmma_f32_16x16x32_bf16(false, a, false, b3,
                                                        (short)0, gacc[3], false, false);
    }
    float bv[4];
#pragma unroll
    for (int j = 0; j < 4; ++j) bv[j] = bias[g * UU + colbase + j * 16 + n];
#pragma unroll
    for (int i = 0; i < 8; ++i) {
      float p = prob[(size_t)g * BB + rowbase + i + 8 * half];
#pragma unroll
      for (int j = 0; j < 4; ++j) facc[j][i] += p * (gacc[j][i] + bv[j]);
    }
  }
#pragma unroll
  for (int i = 0; i < 8; ++i) {
    int row = rowbase + i + 8 * half;
#pragma unroll
    for (int j = 0; j < 4; ++j)
      out[(size_t)row * UU + colbase + j * 16 + n] = facc[j][i];
  }
}

// f32 fallback main GEMM (direct loads + on-the-fly cvt)
__global__ void k_main_f32(const float* __restrict__ h, const float* __restrict__ W,
                           const float* __restrict__ bias, const float* __restrict__ prob,
                           float* __restrict__ out) {
  int lane = threadIdx.x & 31, wave = threadIdx.x >> 5;
  int rowbase = blockIdx.x * 128 + wave * 16;
  int colbase = blockIdx.y * 64;
  int half = lane >> 4, n = lane & 15;
  const float* hrow = h + (size_t)(rowbase + n) * FF;
  v8f facc[4] = {};
  for (int g = 0; g < GG; ++g) {
    const float* Wg = W + (size_t)g * FF * UU;
    v8f gacc[4] = {};
    for (int k0 = 0; k0 < FF; k0 += 32) {
      v16bf a = load_a_f32(hrow, k0, half);
#pragma unroll
      for (int j = 0; j < 4; ++j) {
        v16bf bf = load_b_f32_kstrided(Wg, k0, colbase + j * 16 + n, half);
        gacc[j] = __builtin_amdgcn_wmma_f32_16x16x32_bf16(false, a, false, bf,
                                                          (short)0, gacc[j], false, false);
      }
    }
    float bv[4];
#pragma unroll
    for (int j = 0; j < 4; ++j) bv[j] = bias[g * UU + colbase + j * 16 + n];
#pragma unroll
    for (int i = 0; i < 8; ++i) {
      float p = prob[(size_t)g * BB + rowbase + i + 8 * half];
#pragma unroll
      for (int j = 0; j < 4; ++j) facc[j][i] += p * (gacc[j][i] + bv[j]);
    }
  }
#pragma unroll
  for (int i = 0; i < 8; ++i) {
    int row = rowbase + i + 8 * half;
#pragma unroll
    for (int j = 0; j < 4; ++j)
      out[(size_t)row * UU + colbase + j * 16 + n] = facc[j][i];
  }
}

// ---------------------------------------------------------------------------
extern "C" void kernel_launch(void* const* d_in, const int* in_sizes, int n_in,
                              void* d_out, int out_size, void* d_ws, size_t ws_size,
                              hipStream_t stream) {
  (void)in_sizes; (void)n_in; (void)out_size;
  const float* h    = (const float*)d_in[0];  // [B,F]
  const float* D    = (const float*)d_in[1];  // [G,M,F]
  const float* W    = (const float*)d_in[2];  // [G,F,U]
  const float* bias = (const float*)d_in[3];  // [G,U]
  float* out = (float*)d_out;                 // [B,U]

  // workspace: f32 stats | bf16 h | bf16 D | bf16 W^T
  const size_t STATS_FLOATS = (size_t)BB + GG * MM + GG + 2 * (size_t)GG * BB;
  const size_t STATS_BYTES  = STATS_FLOATS * sizeof(float);   // 16B-aligned
  const size_t HB_ELEMS = (size_t)BB * FF;
  const size_t DB_ELEMS = (size_t)GG * MM * FF;
  const size_t WT_ELEMS = (size_t)GG * UU * FF;
  const size_t NEED = STATS_BYTES + (HB_ELEMS + DB_ELEMS + WT_ELEMS) * 2;

  float* ws   = (float*)d_ws;
  float* h2   = ws;
  float* d2   = h2 + BB;
  float* mkdd = d2 + GG * MM;
  float* mkhd = mkdd + GG;
  float* prob = mkhd + (size_t)GG * BB;

  k_h2<<<BB / 8, 256, 0, stream>>>(h, h2);
  k_dstats<<<GG, 256, 0, stream>>>(D, d2, mkdd);

  if (ws_size >= NEED) {
    __bf16* hb = (__bf16*)((char*)d_ws + STATS_BYTES);
    __bf16* Db = hb + HB_ELEMS;
    __bf16* Wt = Db + DB_ELEMS;

    k_cvt_bf16<<<(int)(HB_ELEMS / 1024), 256, 0, stream>>>(h, hb);
    k_cvt_bf16<<<(int)(DB_ELEMS / 1024), 256, 0, stream>>>(D, Db);
    k_wt<<<dim3(UU / 32, FF / 32, GG), 256, 0, stream>>>(W, Wt);

    k_meankhd_bf16<<<dim3(BB / 128, GG), 256, 0, stream>>>(hb, Db, h2, d2, mkhd);
    k_prob<<<BB / 256, 256, 0, stream>>>(mkhd, mkdd, prob);
    k_main_bf16<<<dim3(BB / 128, UU / 64), 256, 0, stream>>>(hb, Wt, bias, prob, out);
  } else {
    k_meankhd_f32<<<dim3(BB / 128, GG), 256, 0, stream>>>(h, D, h2, d2, mkhd);
    k_prob<<<BB / 256, 256, 0, stream>>>(mkhd, mkdd, prob);
    k_main_f32<<<dim3(BB / 128, UU / 64), 256, 0, stream>>>(h, W, bias, prob, out);
  }
}
